// RNNWithSampling_54425825575650
// MI455X (gfx1250) — compile-verified, matrix operand order checked by reading, not run on hardware
//
#include <hip/hip_runtime.h>
#include <hip/hip_bf16.h>

typedef __attribute__((ext_vector_type(16))) _Float16 v16h;
typedef __attribute__((ext_vector_type(8)))  _Float16 v8h;
typedef __attribute__((ext_vector_type(8)))  float    v8f;

#define BB 16
#define NSTEP 32
#define NSAMPLE 10
#define EMBED 128
#define STATE 512
#define VOCAB 32000
#define NCHAIN (BB * NSAMPLE)      // 160
#define STD_SCALE 0.113f
#define EPSV 1e-5f
#define VTILES (VOCAB / 16)        // 2000

// ---------------------------------------------------------------------------
// logsumexp running-pair merge: (m1,s1) <- merge((m1,s1),(m2,s2))
// ---------------------------------------------------------------------------
__device__ __forceinline__ void lse_merge(float& m1, float& s1, float m2, float s2) {
    if (m2 == -__builtin_inff()) return;
    if (m1 == -__builtin_inff()) { m1 = m2; s1 = s2; return; }
    if (m1 >= m2) { s1 += s2 * expf(m2 - m1); }
    else          { s1 = s1 * expf(m1 - m2) + s2; m1 = m2; }
}

// ---------------------------------------------------------------------------
// Kernel 1: sequential scan. One block per (b,sample) chain (160 blocks).
// ---------------------------------------------------------------------------
__global__ __launch_bounds__(256) void scan_kernel(
    const int* __restrict__ zi, const float* __restrict__ rd,
    const float* __restrict__ latent, const float* __restrict__ Wp,
    const float* __restrict__ bp, const float* __restrict__ sm,
    int* __restrict__ which_out)
{
    const int tid = threadIdx.x;
    const int chain = blockIdx.x;          // 0..159
    const int b = chain / NSAMPLE;

    __shared__ float zs[EMBED], zold[EMBED], zn[EMBED];
    __shared__ float scanbuf[256];
    __shared__ int   s_which;

    if (tid < EMBED) {
        zs[tid]   = latent[(size_t)zi[b] * EMBED + tid];
        zold[tid] = 0.0f;
    }
    __syncthreads();

    for (int t = 0; t < NSTEP; ++t) {
        if (tid == 0) s_which = STATE - 1;

        // ---- std (ddof=1) over 128 elements, then scale ----
        float v = (tid < EMBED) ? zs[tid] : 0.0f;
        scanbuf[tid] = v;
        __syncthreads();
        for (int off = 128; off > 0; off >>= 1) {
            if (tid < off) scanbuf[tid] += scanbuf[tid + off];
            __syncthreads();
        }
        float sum = scanbuf[0];
        __syncthreads();
        scanbuf[tid] = v * v;
        __syncthreads();
        for (int off = 128; off > 0; off >>= 1) {
            if (tid < off) scanbuf[tid] += scanbuf[tid + off];
            __syncthreads();
        }
        float sumsq = scanbuf[0];
        __syncthreads();

        float var   = (sumsq - sum * sum * (1.0f / EMBED)) * (1.0f / (EMBED - 1));
        float scale = STD_SCALE / (EPSV + sqrtf(fmaxf(var, 0.0f)));
        if (tid < EMBED) zn[tid] = zs[tid] * scale;
        __syncthreads();

        // ---- pref[j] = concat(zold, zn) . Wp[j,:] + bp[j], 2 outputs/thread ----
        const int j0 = 2 * tid;
        float p0 = bp[j0], p1 = bp[j0 + 1];
        const float* w0 = Wp + (size_t)j0 * (2 * EMBED);
        const float* w1 = w0 + 2 * EMBED;
        #pragma unroll 4
        for (int d = 0; d < EMBED; ++d) {
            float a = zold[d], c = zn[d];
            p0 += a * w0[d] + c * w0[EMBED + d];
            p1 += a * w1[d] + c * w1[EMBED + d];
        }

        // ---- softmax max-reduce ----
        scanbuf[tid] = fmaxf(p0, p1);
        __syncthreads();
        for (int off = 128; off > 0; off >>= 1) {
            if (tid < off) scanbuf[tid] = fmaxf(scanbuf[tid], scanbuf[tid + off]);
            __syncthreads();
        }
        float mx = scanbuf[0];
        __syncthreads();

        float e0 = expf(p0 - mx), e1 = expf(p1 - mx);
        float partial = e0 + e1;

        // ---- inclusive Hillis-Steele scan over 256 partials ----
        scanbuf[tid] = partial;
        __syncthreads();
        for (int off = 1; off < 256; off <<= 1) {
            float add = (tid >= off) ? scanbuf[tid - off] : 0.0f;
            __syncthreads();
            scanbuf[tid] += add;
            __syncthreads();
        }
        float incl = scanbuf[tid];
        float S    = scanbuf[255];
        float excl = incl - partial;

        // ---- multinomial: first index where cumsum > rd * S ----
        float thr = rd[t * NCHAIN + chain] * S;
        if (excl + e0 > thr)            atomicMin(&s_which, j0);
        else if (excl + e0 + e1 > thr)  atomicMin(&s_which, j0 + 1);
        __syncthreads();
        int w = s_which;
        if (tid == 0) which_out[t * NCHAIN + chain] = w;

        // ---- advance state ----
        if (tid < EMBED) {
            float mv = sm[(size_t)w * EMBED + tid];
            zold[tid] = zn[tid];
            zs[tid]   = zn[tid] + mv;
        }
        __syncthreads();
    }
}

// ---------------------------------------------------------------------------
// Pack kernel: convert a [ntiles*16, 128] f32 matrix into f16 stored in the
// per-lane WMMA operand layout: dst[((tile*32+lane)*64) + ch*16 + hs*8 + e]
// corresponding to src[(tile*16 + lane%16)*128 + ch*32 + (lane/16)*8 + hs*16 + e].
// One block per 16-row tile, 256 threads, each moves 8 contiguous floats.
// ---------------------------------------------------------------------------
__global__ __launch_bounds__(256) void pack_f16_kernel(
    const float* __restrict__ src, _Float16* __restrict__ dst)
{
    const int tile = blockIdx.x;
    const int r    = threadIdx.x;
    const int lane = r >> 3;
    const int q    = r & 7;
    const int ch   = q >> 1;
    const int hs   = q & 1;
    const int row  = tile * 16 + (lane & 15);
    const int d0   = ch * 32 + ((lane >> 4) * 8) + hs * 16;

    const float* s = src + (size_t)row * EMBED + d0;
    v8h o;
    #pragma unroll
    for (int e = 0; e < 8; ++e) o[e] = (_Float16)s[e];
    *(v8h*)(dst + (((size_t)tile * 32 + lane) * 64 + ch * 16 + hs * 8)) = o;
}

// ---------------------------------------------------------------------------
// f32->f16 on-the-fly loaders (used by NM kernel and the fallback LSE kernel)
// ---------------------------------------------------------------------------
__device__ __forceinline__ void load_a_tile(const float* __restrict__ sm,
                                            int ktile, int lane, v16h A[4])
{
    const int m  = lane & 15;
    const int kh = (lane >> 4) * 8;
    const float* arow = sm + (size_t)(ktile * 16 + m) * EMBED;
    #pragma unroll
    for (int ch = 0; ch < 4; ++ch) {
        const int d0 = ch * 32;
        v16h av;
        #pragma unroll
        for (int e = 0; e < 8; ++e) av[e]     = (_Float16)arow[d0 + kh + e];
        #pragma unroll
        for (int e = 0; e < 8; ++e) av[8 + e] = (_Float16)arow[d0 + 16 + kh + e];
        A[ch] = av;
    }
}

__device__ __forceinline__ v16h load_b_chunk(const float* __restrict__ brow,
                                             int ch, int lane)
{
    const int kh = (lane >> 4) * 8;
    const int d0 = ch * 32;
    v16h bv;
    #pragma unroll
    for (int e = 0; e < 8; ++e) bv[e]     = (_Float16)brow[d0 + kh + e];
    #pragma unroll
    for (int e = 0; e < 8; ++e) bv[8 + e] = (_Float16)brow[d0 + 16 + kh + e];
    return bv;
}

// ---------------------------------------------------------------------------
// Kernel 2: NM[k, j] = state_move[k,:] . vocab_W[y[j],:] + vocab_b[y[j]]
// (gathered columns; f32 load + cvt path, tiny: 1024 tiles)
// ---------------------------------------------------------------------------
__global__ __launch_bounds__(64) void nm_kernel(
    const float* __restrict__ sm, const float* __restrict__ vw,
    const float* __restrict__ vb, const int* __restrict__ y,
    float* __restrict__ NM)
{
    const int wave  = threadIdx.x >> 5;
    const int lane  = threadIdx.x & 31;
    const int tile  = blockIdx.x * 2 + wave;   // 0..1023
    const int ktile = tile >> 5;               // 0..31
    const int jtile = tile & 31;               // 0..31

    v16h A[4];
    load_a_tile(sm, ktile, lane, A);

    const int n  = lane & 15;
    const int j  = jtile * 16 + n;
    const int yv = y[j];
    const float* brow = vw + (size_t)yv * EMBED;

    v8f c = {};
    #pragma unroll
    for (int ch = 0; ch < 4; ++ch) {
        v16h bv = load_b_chunk(brow, ch, lane);
        c = __builtin_amdgcn_wmma_f32_16x16x32_f16(false, A[ch], false, bv,
                                                   (short)0, c, false, false);
    }

    const float bias = vb[yv];
    const int hi = lane >> 4;
    #pragma unroll
    for (int r = 0; r < 8; ++r)
        NM[(size_t)(ktile * 16 + r + 8 * hi) * (BB * NSTEP) + j] = c[r] + bias;
}

// ---------------------------------------------------------------------------
// Kernel 3 (fast path): streaming logsumexp using pre-packed f16 operands.
// grid (4 ktile-groups x 50 slabs), 8 waves/block; wave w owns ktile g*8+w.
// All 8 waves stream the SAME vocab addresses -> vocab served once per block.
// Inner loop: 8x global_load_b128 (f16) + 4x v_wmma + online LSE. No cvt.
// ---------------------------------------------------------------------------
__global__ __launch_bounds__(256) void lse_partial_f16_kernel(
    const _Float16* __restrict__ smh, const _Float16* __restrict__ vwh,
    const float* __restrict__ vb, float* __restrict__ slabstats)
{
    const int g     = blockIdx.x;               // 0..3
    const int slab  = blockIdx.y;               // 0..49 (40 vocab tiles each)
    const int wave  = threadIdx.x >> 5;
    const int lane  = threadIdx.x & 31;
    const int ktile = g * 8 + wave;             // 0..31

    // A tile, pre-packed layout: 128 B per lane
    const v16h* ap = (const v16h*)(smh + ((size_t)ktile * 32 + lane) * 64);
    v16h A[4];
    #pragma unroll
    for (int ch = 0; ch < 4; ++ch) A[ch] = ap[ch];

    float mxr[8], sxr[8];
    #pragma unroll
    for (int r = 0; r < 8; ++r) { mxr[r] = -__builtin_inff(); sxr[r] = 0.0f; }

    const int n = lane & 15;
    for (int i = 0; i < 40; ++i) {
        const int vt = slab * 40 + i;
        const v16h* bp = (const v16h*)(vwh + ((size_t)vt * 32 + lane) * 64);

        v8f c = {};
        #pragma unroll
        for (int ch = 0; ch < 4; ++ch)
            c = __builtin_amdgcn_wmma_f32_16x16x32_f16(false, A[ch], false, bp[ch],
                                                       (short)0, c, false, false);

        const float bias = vb[vt * 16 + n];
        #pragma unroll
        for (int r = 0; r < 8; ++r) {
            float logit = c[r] + bias;
            if (logit > mxr[r]) { sxr[r] = sxr[r] * expf(mxr[r] - logit) + 1.0f; mxr[r] = logit; }
            else                { sxr[r] += expf(logit - mxr[r]); }
        }
    }

    // merge across the 16 lanes of each half (columns) via xor shuffles
    #pragma unroll
    for (int mask = 1; mask < 16; mask <<= 1) {
        #pragma unroll
        for (int r = 0; r < 8; ++r) {
            float om = __shfl_xor(mxr[r], mask, 32);
            float os = __shfl_xor(sxr[r], mask, 32);
            lse_merge(mxr[r], sxr[r], om, os);
        }
    }

    if ((lane & 15) == 0) {
        const int rb = (lane >> 4) * 8;         // hi half holds rows 8..15
        #pragma unroll
        for (int r = 0; r < 8; ++r) {
            const int row = ktile * 16 + rb + r;
            slabstats[(size_t)(slab * STATE + row) * 2 + 0] = mxr[r];
            slabstats[(size_t)(slab * STATE + row) * 2 + 1] = sxr[r];
        }
    }
}

// ---------------------------------------------------------------------------
// Kernel 3 (fallback, small workspace): f32 load + cvt in the loop.
// ---------------------------------------------------------------------------
__global__ __launch_bounds__(128) void lse_partial_kernel(
    const float* __restrict__ sm, const float* __restrict__ vw,
    const float* __restrict__ vb, float* __restrict__ slabstats)
{
    const int ktile = blockIdx.x;   // 0..31
    const int slab  = blockIdx.y;   // 0..49
    const int wave  = threadIdx.x >> 5;
    const int lane  = threadIdx.x & 31;

    v16h A[4];
    load_a_tile(sm, ktile, lane, A);

    float mxr[8], sxr[8];
    #pragma unroll
    for (int r = 0; r < 8; ++r) { mxr[r] = -__builtin_inff(); sxr[r] = 0.0f; }

    const int n = lane & 15;
    for (int i = 0; i < 10; ++i) {
        const int vt = slab * 40 + i * 4 + wave;
        const int v  = vt * 16 + n;
        const float* brow = vw + (size_t)v * EMBED;

        v8f c = {};
        #pragma unroll
        for (int ch = 0; ch < 4; ++ch) {
            v16h bv = load_b_chunk(brow, ch, lane);
            c = __builtin_amdgcn_wmma_f32_16x16x32_f16(false, A[ch], false, bv,
                                                       (short)0, c, false, false);
        }
        const float bias = vb[v];
        #pragma unroll
        for (int r = 0; r < 8; ++r) {
            float logit = c[r] + bias;
            if (logit > mxr[r]) { sxr[r] = sxr[r] * expf(mxr[r] - logit) + 1.0f; mxr[r] = logit; }
            else                { sxr[r] += expf(logit - mxr[r]); }
        }
    }

    #pragma unroll
    for (int mask = 1; mask < 16; mask <<= 1) {
        #pragma unroll
        for (int r = 0; r < 8; ++r) {
            float om = __shfl_xor(mxr[r], mask, 32);
            float os = __shfl_xor(sxr[r], mask, 32);
            lse_merge(mxr[r], sxr[r], om, os);
        }
    }

    __shared__ float wm[4][16], wsum[4][16];
    if ((lane & 15) == 0) {
        const int rb = (lane >> 4) * 8;
        #pragma unroll
        for (int r = 0; r < 8; ++r) { wm[wave][rb + r] = mxr[r]; wsum[wave][rb + r] = sxr[r]; }
    }
    __syncthreads();

    if (threadIdx.x < 16) {
        float m = -__builtin_inff(), s = 0.0f;
        #pragma unroll
        for (int w = 0; w < 4; ++w) lse_merge(m, s, wm[w][threadIdx.x], wsum[w][threadIdx.x]);
        const int row = ktile * 16 + threadIdx.x;
        slabstats[(size_t)(slab * STATE + row) * 2 + 0] = m;
        slabstats[(size_t)(slab * STATE + row) * 2 + 1] = s;
    }
}

// ---------------------------------------------------------------------------
// Kernel 4: combine 50 slab pairs -> lse[k]
// ---------------------------------------------------------------------------
__global__ __launch_bounds__(256) void lse_final_kernel(
    const float* __restrict__ slabstats, float* __restrict__ lse)
{
    const int k = blockIdx.x * blockDim.x + threadIdx.x;
    if (k >= STATE) return;
    float m = -__builtin_inff(), s = 0.0f;
    for (int slab = 0; slab < 50; ++slab)
        lse_merge(m, s, slabstats[(size_t)(slab * STATE + k) * 2],
                        slabstats[(size_t)(slab * STATE + k) * 2 + 1]);
    lse[k] = m + logf(s);
}

// ---------------------------------------------------------------------------
// Kernel 5: out[b,t] = mean_s ( NM[which[t,b,s], b*32+t] - lse[which] )
// ---------------------------------------------------------------------------
__global__ __launch_bounds__(256) void out_kernel(
    const int* __restrict__ which, const float* __restrict__ NM,
    const float* __restrict__ lse, float* __restrict__ out)
{
    const int j = blockIdx.x * blockDim.x + threadIdx.x;   // j = b*NSTEP + t
    if (j >= BB * NSTEP) return;
    const int b = j / NSTEP, t = j % NSTEP;
    float acc = 0.0f;
    #pragma unroll
    for (int s = 0; s < NSAMPLE; ++s) {
        int k = which[t * NCHAIN + b * NSAMPLE + s];
        acc += NM[(size_t)k * (BB * NSTEP) + j] - lse[k];
    }
    out[j] = acc * (1.0f / NSAMPLE);
}

// ---------------------------------------------------------------------------
extern "C" void kernel_launch(void* const* d_in, const int* in_sizes, int n_in,
                              void* d_out, int out_size, void* d_ws, size_t ws_size,
                              hipStream_t stream) {
    const int*   zi     = (const int*)  d_in[0];
    const int*   y      = (const int*)  d_in[1];
    const float* rd     = (const float*)d_in[2];
    const float* latent = (const float*)d_in[3];
    const float* Wp     = (const float*)d_in[4];
    const float* bp     = (const float*)d_in[5];
    const float* sm     = (const float*)d_in[6];
    const float* vw     = (const float*)d_in[7];
    const float* vb     = (const float*)d_in[8];
    float* out = (float*)d_out;

    // workspace layout (all offsets 256-aligned)
    constexpr size_t WHICH_OFF = 0;                         // 32*160*4      = 20480
    constexpr size_t NM_OFF    = 20480;                     // 512*512*4     = 1048576
    constexpr size_t SLAB_OFF  = NM_OFF + 1048576;          // 50*512*2*4    = 204800
    constexpr size_t LSE_OFF   = SLAB_OFF + 204800;         // 512*4         = 2048
    constexpr size_t SMH_OFF   = LSE_OFF + 2048;            // 32*32*64*2    = 131072
    constexpr size_t VWH_OFF   = SMH_OFF + 131072;          // 2000*32*64*2  = 8192000
    constexpr size_t NEED_F16  = VWH_OFF + 8192000;         // ~9.6 MB total
    constexpr size_t NEED_MIN  = SMH_OFF;                   // ~1.3 MB

    char* ws = (char*)d_ws;
    int*      which     = (int*)     (ws + WHICH_OFF);
    float*    NM        = (float*)   (ws + NM_OFF);
    float*    slabstats = (float*)   (ws + SLAB_OFF);
    float*    lse       = (float*)   (ws + LSE_OFF);
    _Float16* smh       = (_Float16*)(ws + SMH_OFF);
    _Float16* vwh       = (_Float16*)(ws + VWH_OFF);

    const bool fast = (ws_size >= NEED_F16);
    (void)NEED_MIN;

    // 1) sequential scan -> which indices
    scan_kernel<<<NCHAIN, 256, 0, stream>>>(zi, rd, latent, Wp, bp, sm, which);

    // 2) NM[k, j] for the 512 target vocab rows (WMMA)
    nm_kernel<<<512, 64, 0, stream>>>(sm, vw, vb, y, NM);

    // 3) streaming logsumexp of state_move @ vocab_W.T + vocab_b (WMMA)
    if (fast) {
        pack_f16_kernel<<<STATE / 16, 256, 0, stream>>>(sm, smh);
        pack_f16_kernel<<<VTILES,     256, 0, stream>>>(vw, vwh);
        lse_partial_f16_kernel<<<dim3(4, 50), 256, 0, stream>>>(smh, vwh, vb, slabstats);
    } else {
        lse_partial_kernel<<<dim3(32, 50), 128, 0, stream>>>(sm, vw, vb, slabstats);
    }

    // 4) combine slabs
    lse_final_kernel<<<2, 256, 0, stream>>>(slabstats, lse);

    // 5) gather + mean
    out_kernel<<<2, 256, 0, stream>>>(which, NM, lse, out);
}
